// TransformerBlock_63660005261370
// MI455X (gfx1250) — matrix operand order repaired and checked by reading, not compiled, not run on hardware
//
#include <hip/hip_runtime.h>
#include <cstdint>
#include <cstddef>

// ---------------------------------------------------------------------------
// Transformer block forward for MI455X (gfx1250), bf16 WMMA everywhere.
// ---------------------------------------------------------------------------

#define MODE_F32       0
#define MODE_F32_RES   1
#define MODE_SILU_BF16 2
#define MODE_MUL_BF16  3

typedef __bf16 bf16;
typedef __attribute__((ext_vector_type(16))) __bf16 bf16x16;
typedef __attribute__((ext_vector_type(8)))  __bf16 bf16x8;
typedef __attribute__((ext_vector_type(8)))  float  f32x8;

#define TB_D    1024
#define TB_H    16
#define TB_DK   64
#define TB_FF   4096
#define TB_S    2048
#define TB_B    2
#define TB_NTOK 4096

union FragU { bf16x16 v; bf16x8 h[2]; };

// A-matrix 16x32 bf16 fragment (ISA 7.12.2): row = lane&15;
// elems 0..7 -> K = kh..kh+7, elems 8..15 -> K = kh+16..kh+23, kh = 8*(lane>>4)
__device__ __forceinline__ bf16x16 ld_afrag(const bf16* base, int ld, int lane) {
  const int r  = lane & 15;
  const int kh = (lane >> 4) << 3;
  FragU f;
  f.h[0] = *reinterpret_cast<const bf16x8*>(base + (size_t)r * ld + kh);
  f.h[1] = *reinterpret_cast<const bf16x8*>(base + (size_t)r * ld + kh + 16);
  return f.v;
}

// B-matrix 32x16 bf16 fragment: col = lane&15; elems 0..7 -> K = kb..kb+7,
// elems 8..15 -> K = kb+16..kb+23, kb = 8*(lane>>4).
__device__ __forceinline__ bf16x16 ld_bfrag(const bf16* base, int ld, int lane) {
  const int c  = lane & 15;
  const int kb = (lane >> 4) << 3;
  const bf16* p = base + (size_t)c * ld + kb;
  FragU f;
  f.h[0] = *reinterpret_cast<const bf16x8*>(p);
  f.h[1] = *reinterpret_cast<const bf16x8*>(p + 16);
  return f.v;
}

__device__ __forceinline__ f32x8 wmma_bf16(bf16x16 a, bf16x16 b, f32x8 c) {
  return __builtin_amdgcn_wmma_f32_16x16x32_bf16(false, a, false, b, (short)0, c,
                                                 false, false);
}

__device__ __forceinline__ float red_max16(float x) {
#pragma unroll
  for (int off = 1; off < 16; off <<= 1) x = fmaxf(x, __shfl_xor(x, off, 32));
  return x;
}
__device__ __forceinline__ float red_sum16(float x) {
#pragma unroll
  for (int off = 1; off < 16; off <<= 1) x += __shfl_xor(x, off, 32);
  return x;
}

// ---------------------------------------------------------------------------
// fp32 -> bf16 convert
// ---------------------------------------------------------------------------
__global__ void tb_cvt_bf16(const float* __restrict__ in, bf16* __restrict__ out,
                            int n) {
  int i = blockIdx.x * blockDim.x + threadIdx.x;
  if (i < n) out[i] = (bf16)in[i];
}

// ---------------------------------------------------------------------------
// RMSNorm: one 256-thread block per row of 1024, bf16 output
// ---------------------------------------------------------------------------
__global__ __launch_bounds__(256) void tb_rmsnorm(const float* __restrict__ x,
                                                  const float* __restrict__ w,
                                                  bf16* __restrict__ out) {
  const int row = blockIdx.x;
  const int t = threadIdx.x;
  const float4 xv = reinterpret_cast<const float4*>(x + (size_t)row * TB_D)[t];
  float ss = xv.x * xv.x + xv.y * xv.y + xv.z * xv.z + xv.w * xv.w;
#pragma unroll
  for (int off = 16; off > 0; off >>= 1) ss += __shfl_xor(ss, off, 32);
  __shared__ float red[8];
  if ((t & 31) == 0) red[t >> 5] = ss;
  __syncthreads();
  float tot = 0.f;
#pragma unroll
  for (int i = 0; i < 8; i++) tot += red[i];
  const float s = rsqrtf(tot * (1.0f / TB_D) + 1e-5f);
  const float4 wv = reinterpret_cast<const float4*>(w)[t];
  bf16* o = out + (size_t)row * TB_D + t * 4;
  o[0] = (bf16)(xv.x * s * wv.x);
  o[1] = (bf16)(xv.y * s * wv.y);
  o[2] = (bf16)(xv.z * s * wv.z);
  o[3] = (bf16)(xv.w * s * wv.w);
}

// ---------------------------------------------------------------------------
// RoPE + layout rewrite: qkv fp32 [tok,3072] -> Qr,Kr [b,h,s,d] bf16,
// V -> Vt [b,h,d,s] bf16 (transposed so P*V is an NT WMMA).
// One thread per (b,s,h,pair).
// ---------------------------------------------------------------------------
__global__ __launch_bounds__(256) void tb_rope(const float* __restrict__ qkv,
                                               const int* __restrict__ pos,
                                               bf16* __restrict__ Qr,
                                               bf16* __restrict__ Kr,
                                               bf16* __restrict__ Vt) {
  int tid = blockIdx.x * blockDim.x + threadIdx.x;
  const int i = tid & 31;          // rotary pair index 0..31
  int t2 = tid >> 5;
  const int h = t2 & 15;
  int t3 = t2 >> 4;
  const int s = t3 & (TB_S - 1);
  const int b = t3 >> 11;

  const float p = (float)pos[b * TB_S + s];
  const float freq = __expf(-((float)(2 * i) * (1.0f / TB_DK)) * 9.210340371976184f);
  float sn, cs;
  __sincosf(p * freq, &sn, &cs);

  const float* row = qkv + (size_t)(b * TB_S + s) * (3 * TB_D);
  const int c = h * TB_DK + 2 * i;

  const float qe = row[c], qo = row[c + 1];
  const float ke = row[TB_D + c], ko = row[TB_D + c + 1];
  const float ve = row[2 * TB_D + c], vo = row[2 * TB_D + c + 1];

  const size_t bh = (size_t)(b * TB_H + h);
  bf16* q = Qr + (bh * TB_S + s) * TB_DK + 2 * i;
  bf16* k = Kr + (bh * TB_S + s) * TB_DK + 2 * i;
  q[0] = (bf16)(qe * cs - qo * sn);
  q[1] = (bf16)(qe * sn + qo * cs);
  k[0] = (bf16)(ke * cs - ko * sn);
  k[1] = (bf16)(ke * sn + ko * cs);
  Vt[(bh * TB_DK + 2 * i) * TB_S + s] = (bf16)ve;
  Vt[(bh * TB_DK + 2 * i + 1) * TB_S + s] = (bf16)vo;
}

// ---------------------------------------------------------------------------
// NT GEMM: C[M,N] = A[M,K] * B[N,K]^T, bf16 in, fp32 accum.
// 256 threads = 8 waves arranged 2(M) x 4(N); block tile 128x256;
// wave tile 64x64 (16 WMMAs per 8 fragment loads per K-step of 32).
// Fused epilogues select output form.
// ---------------------------------------------------------------------------
__global__ __launch_bounds__(256) void tb_gemm_nt(
    const bf16* __restrict__ A, const bf16* __restrict__ B,
    float* __restrict__ Cf, bf16* __restrict__ Cb,
    const float* __restrict__ resid, const bf16* __restrict__ mulb,
    int M, int N, int K, int mode) {
  const int lane = threadIdx.x & 31;
  const int wid = threadIdx.x >> 5;
  const int row0 = blockIdx.y * 128 + (wid & 1) * 64;
  const int col0 = blockIdx.x * 256 + (wid >> 1) * 64;

  f32x8 acc[4][4];
#pragma unroll
  for (int i = 0; i < 4; i++)
#pragma unroll
    for (int j = 0; j < 4; j++) acc[i][j] = (f32x8)(0.0f);

  for (int k0 = 0; k0 < K; k0 += 32) {
    bf16x16 af[4], bfr[4];
#pragma unroll
    for (int i = 0; i < 4; i++)
      af[i] = ld_afrag(A + (size_t)(row0 + i * 16) * K + k0, K, lane);
#pragma unroll
    for (int j = 0; j < 4; j++)
      bfr[j] = ld_bfrag(B + (size_t)(col0 + j * 16) * K + k0, K, lane);
#pragma unroll
    for (int i = 0; i < 4; i++)
#pragma unroll
      for (int j = 0; j < 4; j++) acc[i][j] = wmma_bf16(af[i], bfr[j], acc[i][j]);
  }

  const int hi = lane >> 4, cn = lane & 15;
#pragma unroll
  for (int i = 0; i < 4; i++)
#pragma unroll
    for (int j = 0; j < 4; j++)
#pragma unroll
      for (int v = 0; v < 8; v++) {
        const int m = row0 + i * 16 + v + 8 * hi;
        const int n = col0 + j * 16 + cn;
        const size_t idx = (size_t)m * N + n;
        const float val = acc[i][j][v];
        if (mode == MODE_F32) {
          Cf[idx] = val;
        } else if (mode == MODE_F32_RES) {
          Cf[idx] = val + resid[idx];
        } else if (mode == MODE_SILU_BF16) {
          Cb[idx] = (bf16)(val / (1.0f + __expf(-val)));
        } else {  // MODE_MUL_BF16
          Cb[idx] = (bf16)(val * (float)mulb[idx]);
        }
      }
}

// ---------------------------------------------------------------------------
// Causal flash attention. Grid: (S/128, B*H), 128 threads (4 waves).
// Each wave owns 32 query rows; online softmax in WMMA C-layout;
// P staged through per-wave LDS tile to become an A-fragment for P*V.
// ---------------------------------------------------------------------------
__global__ __launch_bounds__(128) void tb_attn(const bf16* __restrict__ Qr,
                                               const bf16* __restrict__ Kr,
                                               const bf16* __restrict__ Vt,
                                               bf16* __restrict__ Out) {
  __shared__ __align__(16) bf16 pLds[4][32][32];

  const int lane = threadIdx.x & 31;
  const int wid = threadIdx.x >> 5;
  const int bh = blockIdx.y;
  const int h = bh & (TB_H - 1);
  const int b = bh >> 4;
  const int qRow0 = blockIdx.x * 128 + wid * 32;

  const bf16* Qp = Qr + ((size_t)bh * TB_S + qRow0) * TB_DK;
  const bf16* Kp = Kr + (size_t)bh * TB_S * TB_DK;
  const bf16* Vp = Vt + (size_t)bh * TB_DK * TB_S;

  bf16x16 qf[2][2];
#pragma unroll
  for (int i = 0; i < 2; i++)
#pragma unroll
    for (int kc = 0; kc < 2; kc++)
      qf[i][kc] = ld_afrag(Qp + (size_t)(i * 16) * TB_DK + kc * 32, TB_DK, lane);

  f32x8 o[2][4];
#pragma unroll
  for (int i = 0; i < 2; i++)
#pragma unroll
    for (int d = 0; d < 4; d++) o[i][d] = (f32x8)(0.0f);
  f32x8 mrow[2], lrow[2];
#pragma unroll
  for (int i = 0; i < 2; i++) { mrow[i] = (f32x8)(-1e30f); lrow[i] = (f32x8)(0.0f); }

  const int hi = lane >> 4, cn = lane & 15;
  const int kbEnd = blockIdx.x * 128 + 128;  // uniform across block

  for (int kb = 0; kb < kbEnd; kb += 32) {
    // --- scores S = (Q K^T) * scale, causal mask -----------------------
    bf16x16 kf[2][2];
#pragma unroll
    for (int j = 0; j < 2; j++)
#pragma unroll
      for (int kc = 0; kc < 2; kc++)
        kf[j][kc] = ld_bfrag(Kp + (size_t)(kb + j * 16) * TB_DK + kc * 32,
                             TB_DK, lane);
    f32x8 sc[2][2];
#pragma unroll
    for (int i = 0; i < 2; i++)
#pragma unroll
      for (int j = 0; j < 2; j++) {
        f32x8 z = (f32x8)(0.0f);
        z = wmma_bf16(qf[i][0], kf[j][0], z);
        sc[i][j] = wmma_bf16(qf[i][1], kf[j][1], z);
      }
#pragma unroll
    for (int i = 0; i < 2; i++)
#pragma unroll
      for (int j = 0; j < 2; j++)
#pragma unroll
        for (int v = 0; v < 8; v++) {
          const int gRow = qRow0 + i * 16 + v + 8 * hi;
          const int gCol = kb + j * 16 + cn;
          float val = sc[i][j][v] * 0.125f;  // 1/sqrt(64)
          if (gCol > gRow) val = -1e30f;
          sc[i][j][v] = val;
        }

    // --- online softmax per 16-row tile -------------------------------
#pragma unroll
    for (int i = 0; i < 2; i++) {
      f32x8 mnew, scale;
#pragma unroll
      for (int v = 0; v < 8; v++) {
        float t = red_max16(fmaxf(sc[i][0][v], sc[i][1][v]));
        mnew[v] = fmaxf(mrow[i][v], t);
        scale[v] = __expf(mrow[i][v] - mnew[v]);
        mrow[i][v] = mnew[v];
      }
#pragma unroll
      for (int j = 0; j < 2; j++)
#pragma unroll
        for (int v = 0; v < 8; v++)
          sc[i][j][v] = __expf(sc[i][j][v] - mnew[v]);
#pragma unroll
      for (int v = 0; v < 8; v++) {
        const float rs = red_sum16(sc[i][0][v] + sc[i][1][v]);
        lrow[i][v] = lrow[i][v] * scale[v] + rs;
      }
#pragma unroll
      for (int d = 0; d < 4; d++)
#pragma unroll
        for (int v = 0; v < 8; v++) o[i][d][v] *= scale[v];
      // stage P tile (C-layout -> LDS row-major) for A-fragment reload
#pragma unroll
      for (int j = 0; j < 2; j++)
#pragma unroll
        for (int v = 0; v < 8; v++)
          pLds[wid][i * 16 + v + 8 * hi][j * 16 + cn] = (bf16)sc[i][j][v];
    }
    __syncthreads();

    // --- O += P * V ----------------------------------------------------
    bf16x16 pf[2];
#pragma unroll
    for (int i = 0; i < 2; i++)
      pf[i] = ld_afrag(&pLds[wid][i * 16][0], 32, lane);
    bf16x16 vf[4];
#pragma unroll
    for (int d = 0; d < 4; d++)
      vf[d] = ld_bfrag(Vp + (size_t)(d * 16) * TB_S + kb, TB_S, lane);
#pragma unroll
    for (int i = 0; i < 2; i++)
#pragma unroll
      for (int d = 0; d < 4; d++) o[i][d] = wmma_bf16(pf[i], vf[d], o[i][d]);
    __syncthreads();
  }

  // --- finalize: O /= l, write [tok, h*64+d] bf16 ----------------------
#pragma unroll
  for (int i = 0; i < 2; i++)
#pragma unroll
    for (int v = 0; v < 8; v++) {
      const float rinv = 1.0f / fmaxf(lrow[i][v], 1e-30f);
      const int tok = b * TB_S + qRow0 + i * 16 + v + 8 * hi;
#pragma unroll
      for (int d = 0; d < 4; d++)
        Out[(size_t)tok * TB_D + h * TB_DK + d * 16 + cn] =
            (bf16)(o[i][d][v] * rinv);
    }
}

// ---------------------------------------------------------------------------
// Host orchestration
// ---------------------------------------------------------------------------
extern "C" void kernel_launch(void* const* d_in, const int* in_sizes, int n_in,
                              void* d_out, int out_size, void* d_ws,
                              size_t ws_size, hipStream_t stream) {
  const float* x = (const float*)d_in[0];
  const int* pos = (const int*)d_in[1];
  const float* n1w = (const float*)d_in[2];
  const float* n2w = (const float*)d_in[3];
  const float* wq = (const float*)d_in[4];
  const float* wk = (const float*)d_in[5];
  const float* wv = (const float*)d_in[6];
  const float* wo = (const float*)d_in[7];
  const float* w1 = (const float*)d_in[8];
  const float* w2 = (const float*)d_in[9];
  const float* w3 = (const float*)d_in[10];
  float* out = (float*)d_out;
  char* ws = (char*)d_ws;

  // Workspace layout (bytes); regions R1/R2 are reused phase-wise.
  bf16* Wqkv = (bf16*)(ws + 0);            //  6 MB: [wq;wk;wv] bf16 (3072x1024)
  bf16* Wo   = (bf16*)(ws + 6291456);      //  2 MB
  bf16* W1   = (bf16*)(ws + 8388608);      //  8 MB
  bf16* W3   = (bf16*)(ws + 16777216);     //  8 MB
  bf16* W2   = (bf16*)(ws + 25165824);     //  8 MB
  bf16* Xn   = (bf16*)(ws + 33554432);     //  8 MB (xn1, later xn2)
  float* QKV = (float*)(ws + 41943040);    // 48 MB fp32 (R1)
  bf16* Tbuf = (bf16*)(ws + 41943040);     //   (R1 reuse) silu(a) 32 MB
  bf16* Qr   = (bf16*)(ws + 92274688);     //  8 MB (R2)
  bf16* Kr   = (bf16*)(ws + 100663296);    //  8 MB (R2)
  bf16* Vt   = (bf16*)(ws + 109051904);    //  8 MB (R2)
  bf16* Attn = (bf16*)(ws + 117440512);    //  8 MB (R2)
  bf16* Hbuf = (bf16*)(ws + 92274688);     //   (R2 reuse) h 32 MB
  float* H1  = (float*)(ws + 125829120);   // 16 MB fp32; total 142.6 MB

  const int DD = TB_D * TB_D;        // 1048576
  const int FD = TB_FF * TB_D;       // 4194304

  // 1) weights -> bf16
  tb_cvt_bf16<<<(DD + 255) / 256, 256, 0, stream>>>(wq, Wqkv, DD);
  tb_cvt_bf16<<<(DD + 255) / 256, 256, 0, stream>>>(wk, Wqkv + DD, DD);
  tb_cvt_bf16<<<(DD + 255) / 256, 256, 0, stream>>>(wv, Wqkv + 2 * DD, DD);
  tb_cvt_bf16<<<(DD + 255) / 256, 256, 0, stream>>>(wo, Wo, DD);
  tb_cvt_bf16<<<(FD + 255) / 256, 256, 0, stream>>>(w1, W1, FD);
  tb_cvt_bf16<<<(FD + 255) / 256, 256, 0, stream>>>(w3, W3, FD);
  tb_cvt_bf16<<<(FD + 255) / 256, 256, 0, stream>>>(w2, W2, FD);

  // 2) xn1 = rmsnorm(x, norm1_w)
  tb_rmsnorm<<<TB_NTOK, 256, 0, stream>>>(x, n1w, Xn);

  // 3) qkv = xn1 @ [wq;wk;wv]^T  (M=4096, N=3072, K=1024) fp32
  tb_gemm_nt<<<dim3(3072 / 256, TB_NTOK / 128), 256, 0, stream>>>(
      Xn, Wqkv, QKV, nullptr, nullptr, nullptr, TB_NTOK, 3 * TB_D, TB_D, MODE_F32);

  // 4) RoPE + head layout (Qr,Kr [b,h,s,d]; Vt [b,h,d,s]) bf16
  tb_rope<<<(TB_B * TB_S * TB_H * 32) / 256, 256, 0, stream>>>(QKV, pos, Qr, Kr, Vt);

  // 5) causal flash attention -> Attn [tok, 1024] bf16
  tb_attn<<<dim3(TB_S / 128, TB_B * TB_H), 128, 0, stream>>>(Qr, Kr, Vt, Attn);

  // 6) h1 = x + Attn @ wo^T  (fp32)
  tb_gemm_nt<<<dim3(TB_D / 256, TB_NTOK / 128), 256, 0, stream>>>(
      Attn, Wo, H1, nullptr, x, nullptr, TB_NTOK, TB_D, TB_D, MODE_F32_RES);

  // 7) xn2 = rmsnorm(h1, norm2_w)
  tb_rmsnorm<<<TB_NTOK, 256, 0, stream>>>(H1, n2w, Xn);

  // 8) t = silu(xn2 @ w1^T)  bf16
  tb_gemm_nt<<<dim3(TB_FF / 256, TB_NTOK / 128), 256, 0, stream>>>(
      Xn, W1, nullptr, Tbuf, nullptr, nullptr, TB_NTOK, TB_FF, TB_D, MODE_SILU_BF16);

  // 9) h = t * (xn2 @ w3^T)  bf16
  tb_gemm_nt<<<dim3(TB_FF / 256, TB_NTOK / 128), 256, 0, stream>>>(
      Xn, W3, nullptr, Hbuf, nullptr, Tbuf, TB_NTOK, TB_FF, TB_D, MODE_MUL_BF16);

  // 10) out = h1 + h @ w2^T  (fp32, K=4096)
  tb_gemm_nt<<<dim3(TB_D / 256, TB_NTOK / 128), 256, 0, stream>>>(
      Hbuf, W2, out, nullptr, H1, nullptr, TB_NTOK, TB_D, TB_FF, MODE_F32_RES);
}
